// PatchCore_78821239816348
// MI455X (gfx1250) — matrix operand order repaired and checked by reading, not compiled, not run on hardware
//
#include <hip/hip_runtime.h>
#include <cstddef>

// ---------------------------------------------------------------------------
// PatchCore scoring on MI455X (gfx1250, wave32, WMMA).
// Main GEMM uses v_wmma_f32_16x16x32_bf16; norms kept in f32.
// patch is pre-converted to bf16 once; patch_lib streams from HBM once.
// ---------------------------------------------------------------------------

typedef __attribute__((ext_vector_type(16))) __bf16 v16bf;
typedef __attribute__((ext_vector_type(8)))  float  v8f;

#define PC_M        784
#define PC_N        200000
#define PC_K        384
#define PC_MT       49           // 784 / 16 row tiles
#define PC_KS       12           // 384 / 32 K-steps
#define GEMM_BLOCKS 1563         // ceil(200000 / 128)
#define WD_BLOCKS   196
#define OUT_SIZE    224

// workspace byte offsets
#define WS_ROWKEYS  0            // 784 * 8  = 6272
#define WS_A2       6400         // 784 * 4  = 3136
#define WS_MINVAL   9600         // 784 * 4
#define WS_MINIDX   12800        // 784 * 4
#define WS_SCAL     16000        // [0]=s_star(f32) [1]=s_idx(i32) [2]=m_star idx(i32)
#define WS_TOP3     16064        // 196 * 3 * 8 = 4704 -> ends 20768
#define WS_ABF16    20800        // 784 * 384 * 2 = 602112 (16B aligned)

union BF16Frag {
    v16bf        v;
    unsigned int u[8];
};

__device__ __forceinline__ unsigned int f32_to_bf16(float f) {
    unsigned int u = __float_as_uint(f);
    u += 0x7FFFu + ((u >> 16) & 1u);   // round-to-nearest-even
    return u >> 16;
}
__device__ __forceinline__ unsigned int bf16pk(float lo, float hi) {
    return f32_to_bf16(lo) | (f32_to_bf16(hi) << 16);
}
__device__ __forceinline__ unsigned long long shfl_xor_u64(unsigned long long v, int mask) {
    unsigned int lo = (unsigned int)v, hi = (unsigned int)(v >> 32);
    lo = __shfl_xor(lo, mask, 32);
    hi = __shfl_xor(hi, mask, 32);
    return ((unsigned long long)hi << 32) | lo;
}
__device__ __forceinline__ unsigned long long enc_key(float d2, unsigned int idx) {
    // d2 >= 0 -> float bits are order-preserving as uint
    return ((unsigned long long)__float_as_uint(d2) << 32) | idx;
}

// --------------------------------------------------------------------------
// 0) init: row-min keys to +inf, a2[i] = |patch[i]|^2 (exact f32), and a
//    one-time f32->bf16 conversion of patch (so the GEMM never converts A).
// --------------------------------------------------------------------------
__global__ void pc_init(const float* __restrict__ patch,
                        unsigned long long* __restrict__ rowkeys,
                        float* __restrict__ a2,
                        unsigned int* __restrict__ abf16_u32) {
    int i = blockIdx.x * blockDim.x + threadIdx.x;
    if (i < PC_M) {
        rowkeys[i] = ~0ull;
        const float* pr = patch + (size_t)i * PC_K;
        unsigned int* dst = abf16_u32 + (size_t)i * (PC_K / 2);
        float s = 0.f;
        #pragma unroll 4
        for (int k = 0; k < PC_K; k += 2) {
            float f0 = pr[k], f1 = pr[k + 1];
            s = fmaf(f0, f0, s);
            s = fmaf(f1, f1, s);
            dst[k >> 1] = bf16pk(f0, f1);
        }
        a2[i] = s;
    }
}

// --------------------------------------------------------------------------
// 1) main GEMM + row-min. Grid: 1563 blocks x 256 threads (8 waves).
//    Wave w owns 16 library columns; B fragments (12 x v16bf) live in VGPRs
//    for the whole block so patch_lib streams from HBM exactly once.
//    For each of 49 M-tiles: copy pre-converted bf16 A tile (16x384) into
//    LDS (pure uint4 copy), 12 x v_wmma_f32_16x16x32_bf16, then
//    d2 = a2 + b2 - 2*dot and keyed row-min.
// --------------------------------------------------------------------------
__global__ __launch_bounds__(256) void pc_gemm_min(
        const unsigned short* __restrict__ abf16, const float* __restrict__ lib,
        const float* __restrict__ a2, unsigned long long* __restrict__ rowkeys) {
    __shared__ __align__(16) unsigned short lds_a[16 * PC_K];   // 12 KB bf16 A tile
    __shared__ unsigned long long smin[16];

    const int tid  = threadIdx.x;
    const int wave = tid >> 5;
    const int lane = tid & 31;
    const int l15  = lane & 15;          // A row within tile / B column within group
    const int kq   = lane >> 4;          // lane half selector
    const int kh8  = kq << 3;            // A K-offset (0/8) and C row offset (0/8)
    const int kh16 = kq << 4;            // unused; B K-offset is kh8 like A

    const int  j  = blockIdx.x * 128 + wave * 16 + l15;
    const bool jv = (j < PC_N);
    const size_t jj = jv ? (size_t)j : (size_t)(PC_N - 1);
    const float* bp = lib + jj * PC_K;

    // ---- load B fragments for all K once; accumulate |b|^2 in f32 ----
    BF16Frag bfrag[PC_KS];
    float bsq = 0.f;
    #pragma unroll
    for (int s = 0; s < PC_KS; ++s) {
        const float* q = bp + s * 32 + kh8;           // K = kh8..+7 and 16+kh8..+7
        float4 f0 = *(const float4*)(q +  0);
        float4 f1 = *(const float4*)(q +  4);
        float4 f2 = *(const float4*)(q + 16);
        float4 f3 = *(const float4*)(q + 20);
        bsq += f0.x*f0.x + f0.y*f0.y + f0.z*f0.z + f0.w*f0.w;
        bsq += f1.x*f1.x + f1.y*f1.y + f1.z*f1.z + f1.w*f1.w;
        bsq += f2.x*f2.x + f2.y*f2.y + f2.z*f2.z + f2.w*f2.w;
        bsq += f3.x*f3.x + f3.y*f3.y + f3.z*f3.z + f3.w*f3.w;
        bfrag[s].u[0] = bf16pk(f0.x, f0.y);  bfrag[s].u[1] = bf16pk(f0.z, f0.w);
        bfrag[s].u[2] = bf16pk(f1.x, f1.y);  bfrag[s].u[3] = bf16pk(f1.z, f1.w);
        bfrag[s].u[4] = bf16pk(f2.x, f2.y);  bfrag[s].u[5] = bf16pk(f2.z, f2.w);
        bfrag[s].u[6] = bf16pk(f3.x, f3.y);  bfrag[s].u[7] = bf16pk(f3.z, f3.w);
    }
    const float b2j = bsq + __shfl_xor(bsq, 16, 32);  // combine K halves

    // ---- march over all 49 row tiles; bf16 patch copy is L2-resident ----
    for (int mt = 0; mt < PC_MT; ++mt) {
        const int mbase = mt * 16;

        if (tid < 16) smin[tid] = ~0ull;
        {   // pure copy staging: 16*384 bf16 = 768 uint4, 3 per thread
            const uint4* src = (const uint4*)(abf16 + (size_t)mbase * PC_K);
            uint4* dst = (uint4*)lds_a;
            #pragma unroll
            for (int c = 0; c < 3; ++c) dst[tid + 256 * c] = src[tid + 256 * c];
        }
        __syncthreads();

        v8f acc = {0.f, 0.f, 0.f, 0.f, 0.f, 0.f, 0.f, 0.f};
        #pragma unroll
        for (int s = 0; s < PC_KS; ++s) {
            const int kb = s * 32;
            const unsigned short* ap = &lds_a[l15 * PC_K + kb + kh8];
            uint4 q0 = *(const uint4*)(ap);        // K = kb+kh8   .. +7
            uint4 q1 = *(const uint4*)(ap + 16);   // K = kb+16+kh8.. +7
            BF16Frag af;
            af.u[0] = q0.x; af.u[1] = q0.y; af.u[2] = q0.z; af.u[3] = q0.w;
            af.u[4] = q1.x; af.u[5] = q1.y; af.u[6] = q1.z; af.u[7] = q1.w;
            acc = __builtin_amdgcn_wmma_f32_16x16x32_bf16(
                      false, af.v, false, bfrag[s].v, (short)0, acc, false, false);
        }

        // d2 = a2 + b2 - 2*dot; keyed row-min (value<<32 | col idx)
        unsigned long long runkey[8];
        #pragma unroll
        for (int r = 0; r < 8; ++r) {
            float d2 = a2[mbase + r + kh8] + b2j - 2.0f * acc[r];
            d2 = fmaxf(d2, 0.0f);
            unsigned long long key = jv ? enc_key(d2, (unsigned int)j) : ~0ull;
            #pragma unroll
            for (int msk = 1; msk <= 8; msk <<= 1) {
                unsigned long long o = shfl_xor_u64(key, msk);
                if (o < key) key = o;
            }
            runkey[r] = key;
        }
        if (l15 == 0) {
            #pragma unroll
            for (int r = 0; r < 8; ++r)
                atomicMin(&smin[r + kh8], runkey[r]);      // ds_min_u64
        }
        __syncthreads();
        if (tid < 16) atomicMin(&rowkeys[mbase + tid], smin[tid]);  // global_atomic_min_u64
        __syncthreads();
    }
}

// --------------------------------------------------------------------------
// 2) finalize row mins: min_val = sqrt(d2), argmax -> s_star / s_idx / m_star
// --------------------------------------------------------------------------
__global__ __launch_bounds__(256) void pc_finalize_min(
        const unsigned long long* __restrict__ rowkeys,
        float* __restrict__ minval, int* __restrict__ minidx,
        float* __restrict__ scal, int* __restrict__ iscal) {
    __shared__ int   sidxs[PC_M];
    __shared__ float rmax[256];
    __shared__ int   rmi[256];
    const int tid = threadIdx.x;
    float best = -1.f; int bi = 0;
    for (int i = tid; i < PC_M; i += 256) {
        unsigned long long k = rowkeys[i];
        float d2 = __uint_as_float((unsigned int)(k >> 32));
        float mv = sqrtf(fmaxf(d2, 0.f));
        int   mi = (int)(unsigned int)k;
        minval[i] = mv; minidx[i] = mi; sidxs[i] = mi;
        if (mv > best || (mv == best && i < bi)) { best = mv; bi = i; }
    }
    rmax[tid] = best; rmi[tid] = bi;
    __syncthreads();
    for (int s = 128; s > 0; s >>= 1) {
        if (tid < s) {
            if (rmax[tid + s] > rmax[tid] ||
                (rmax[tid + s] == rmax[tid] && rmi[tid + s] < rmi[tid])) {
                rmax[tid] = rmax[tid + s]; rmi[tid] = rmi[tid + s];
            }
        }
        __syncthreads();
    }
    if (tid == 0) {
        scal[0]  = rmax[0];
        iscal[1] = rmi[0];
        int mi = (rmi[0] >= 0 && rmi[0] < PC_M) ? sidxs[rmi[0]] : 0;
        iscal[2] = (mi >= 0 && mi < PC_N) ? mi : 0;
    }
}

// --------------------------------------------------------------------------
// 3) w_dist = cdist(m_star, lib); per-block top-3 smallest (keyed)
// --------------------------------------------------------------------------
__global__ __launch_bounds__(256) void pc_wdist_top3(
        const float* __restrict__ lib, const int* __restrict__ iscal,
        unsigned long long* __restrict__ top3) {
    __shared__ float mv[PC_K];
    __shared__ unsigned long long sh[256 * 3];
    const int tid = threadIdx.x;
    int mstar = iscal[2];
    if (mstar < 0 || mstar >= PC_N) mstar = 0;
    for (int k = tid; k < PC_K; k += 256) mv[k] = lib[(size_t)mstar * PC_K + k];
    __syncthreads();

    unsigned long long t0 = ~0ull, t1 = ~0ull, t2 = ~0ull;
    for (int j = blockIdx.x * 256 + tid; j < PC_N; j += WD_BLOCKS * 256) {
        const float* r = lib + (size_t)j * PC_K;
        float d2 = 0.f;
        #pragma unroll 4
        for (int k = 0; k < PC_K; ++k) { float d = r[k] - mv[k]; d2 = fmaf(d, d, d2); }
        unsigned long long key = enc_key(d2, (unsigned int)j);
        if (key < t0)      { t2 = t1; t1 = t0; t0 = key; }
        else if (key < t1) { t2 = t1; t1 = key; }
        else if (key < t2) { t2 = key; }
    }
    sh[tid * 3 + 0] = t0; sh[tid * 3 + 1] = t1; sh[tid * 3 + 2] = t2;
    __syncthreads();
    if (tid == 0) {
        unsigned long long b0 = ~0ull, b1 = ~0ull, b2 = ~0ull;
        for (int i = 0; i < 256 * 3; ++i) {
            unsigned long long k = sh[i];
            if (k < b0)      { b2 = b1; b1 = b0; b0 = k; }
            else if (k < b1) { b2 = b1; b1 = k; }
            else if (k < b2) { b2 = k; }
        }
        top3[blockIdx.x * 3 + 0] = b0;
        top3[blockIdx.x * 3 + 1] = b1;
        top3[blockIdx.x * 3 + 2] = b2;
    }
}

// --------------------------------------------------------------------------
// 4) merge partial top-3, compute reweighted score s -> d_out[0]
// --------------------------------------------------------------------------
__global__ void pc_final_score(const float* __restrict__ patch,
                               const float* __restrict__ lib,
                               const float* __restrict__ scal,
                               const int* __restrict__ iscal,
                               const unsigned long long* __restrict__ top3,
                               float* __restrict__ out) {
    if (threadIdx.x != 0 || blockIdx.x != 0) return;
    unsigned long long b0 = ~0ull, b1 = ~0ull, b2 = ~0ull;
    for (int i = 0; i < WD_BLOCKS * 3; ++i) {
        unsigned long long k = top3[i];
        if (k < b0)      { b2 = b1; b1 = b0; b0 = k; }
        else if (k < b1) { b2 = b1; b1 = k; }
        else if (k < b2) { b2 = k; }
    }
    int i1 = (int)(unsigned int)b1;   // nn_idx[1]
    int i2 = (int)(unsigned int)b2;   // nn_idx[2]
    if (i1 < 0 || i1 >= PC_N) i1 = 0;
    if (i2 < 0 || i2 >= PC_N) i2 = 0;
    int sidx = iscal[1];
    if (sidx < 0 || sidx >= PC_M) sidx = 0;

    const float* mt = patch + (size_t)sidx * PC_K;
    const float* r1 = lib + (size_t)i1 * PC_K;
    const float* r2 = lib + (size_t)i2 * PC_K;
    float n1 = 0.f, n2 = 0.f;
    for (int k = 0; k < PC_K; ++k) {
        float d1 = mt[k] - r1[k]; n1 = fmaf(d1, d1, n1);
        float d2 = mt[k] - r2[k]; n2 = fmaf(d2, d2, n2);
    }
    n1 = sqrtf(n1); n2 = sqrtf(n2);
    const float Dsq = 19.59591794f;   // sqrt(384)
    float s_star = scal[0];
    float w = 1.0f - expf(s_star / Dsq) / (expf(n1 / Dsq) + expf(n2 / Dsq));
    out[0] = w * s_star;
}

// --------------------------------------------------------------------------
// 5) bilinear resize 28x28 -> 224x224 (half-pixel centers, edge clamp)
// --------------------------------------------------------------------------
__global__ void pc_resize(const float* __restrict__ minval, float* __restrict__ out) {
    int p = blockIdx.x * blockDim.x + threadIdx.x;
    if (p >= OUT_SIZE * OUT_SIZE) return;
    int y = p / OUT_SIZE, x = p - y * OUT_SIZE;
    float sy = (y + 0.5f) * 0.125f - 0.5f;
    float sx = (x + 0.5f) * 0.125f - 0.5f;
    float fy = floorf(sy), fx = floorf(sx);
    int y0 = (int)fy, x0 = (int)fx;
    float wy = sy - fy, wx = sx - fx;
    int y0c = min(max(y0, 0), 27), y1c = min(max(y0 + 1, 0), 27);
    int x0c = min(max(x0, 0), 27), x1c = min(max(x0 + 1, 0), 27);
    float v00 = minval[y0c * 28 + x0c], v01 = minval[y0c * 28 + x1c];
    float v10 = minval[y1c * 28 + x0c], v11 = minval[y1c * 28 + x1c];
    out[1 + p] = (1.f - wy) * ((1.f - wx) * v00 + wx * v01)
               +        wy  * ((1.f - wx) * v10 + wx * v11);
}

// --------------------------------------------------------------------------
extern "C" void kernel_launch(void* const* d_in, const int* in_sizes, int n_in,
                              void* d_out, int out_size, void* d_ws, size_t ws_size,
                              hipStream_t stream) {
    (void)in_sizes; (void)n_in; (void)out_size; (void)ws_size;
    const float* patch = (const float*)d_in[0];
    const float* lib   = (const float*)d_in[1];
    char* ws = (char*)d_ws;
    unsigned long long* rowkeys = (unsigned long long*)(ws + WS_ROWKEYS);
    float*              a2      = (float*)(ws + WS_A2);
    float*              minval  = (float*)(ws + WS_MINVAL);
    int*                minidx  = (int*)(ws + WS_MINIDX);
    float*              scal    = (float*)(ws + WS_SCAL);
    int*                iscal   = (int*)(ws + WS_SCAL);
    unsigned long long* top3    = (unsigned long long*)(ws + WS_TOP3);
    unsigned short*     abf16   = (unsigned short*)(ws + WS_ABF16);
    float*              out     = (float*)d_out;

    pc_init<<<(PC_M + 255) / 256, 256, 0, stream>>>(patch, rowkeys, a2,
                                                    (unsigned int*)abf16);
    pc_gemm_min<<<GEMM_BLOCKS, 256, 0, stream>>>(abf16, lib, a2, rowkeys);
    pc_finalize_min<<<1, 256, 0, stream>>>(rowkeys, minval, minidx, scal, iscal);
    pc_wdist_top3<<<WD_BLOCKS, 256, 0, stream>>>(lib, iscal, top3);
    pc_final_score<<<1, 32, 0, stream>>>(patch, lib, scal, iscal, top3, out);
    pc_resize<<<WD_BLOCKS, 256, 0, stream>>>(minval, out);
}
